// BridgeNetUp_knn_37855841747273
// MI455X (gfx1250) — compile-verified, matrix operand order checked by reading, not run on hardware
//
#include <hip/hip_runtime.h>
#include <hip/hip_bf16.h>
#include <stdint.h>

// ---------------------------------------------------------------------------
// Problem constants (from reference setup_inputs)
// ---------------------------------------------------------------------------
#define BB   16
#define SS   1024
#define NN   4096
#define C1   256
#define C2   128
#define CIN  384          // C1 + C2
#define HH   256
#define MM   (BB * NN)    // 65536 rows
#define BN_EPS 1e-5f

typedef __attribute__((ext_vector_type(16))) __bf16 v16bf;
typedef __attribute__((ext_vector_type(8)))  __bf16 v8bf;
typedef __attribute__((ext_vector_type(8)))  float  v8f;

// ---------------------------------------------------------------------------
// 1) f32 -> bf16 conversion (weights)
// ---------------------------------------------------------------------------
__global__ __launch_bounds__(256) void f32_to_bf16_kernel(const float* __restrict__ in,
                                                          __bf16* __restrict__ out, int n) {
    int i = blockIdx.x * 256 + threadIdx.x;
    if (i < n) out[i] = (__bf16)in[i];
}

// ---------------------------------------------------------------------------
// 2) zero helper for stats buffers
// ---------------------------------------------------------------------------
__global__ __launch_bounds__(256) void zero_f32_kernel(float* __restrict__ p, int n) {
    int i = blockIdx.x * 256 + threadIdx.x;
    if (i < n) p[i] = 0.0f;
}

// ---------------------------------------------------------------------------
// 3) KNN (k=3) inverse-distance interpolate + concat -> X [M, CIN] bf16
//    grid = (NN/256, BB), block = 256.
//    xyz1 for the batch staged into LDS with the CDNA5 async global->LDS
//    path (global_load_async_to_lds_b128, ASYNCcnt), no VGPR round-trip.
// ---------------------------------------------------------------------------
__global__ __launch_bounds__(256) void knn_build_x_kernel(
    const float* __restrict__ points1,   // [B,S,C1]
    const float* __restrict__ points2,   // [B,N,C2]
    const float* __restrict__ xyz1,      // [B,S,3]
    const float* __restrict__ xyz2,      // [B,N,3]
    __bf16* __restrict__ X)              // [M, CIN]
{
    __shared__ float sxyz[SS * 3];       // 12 KB of the 320 KB WGP LDS
    const int b   = blockIdx.y;
    const int tid = threadIdx.x;

    // Async-stage xyz1[b] (12288 B) into LDS: 768 x b128, 3 per lane.
    {
        const float* x1 = xyz1 + (size_t)b * SS * 3;
        #pragma unroll
        for (int it = 0; it < 3; ++it) {
            const int i = tid + it * 256;                      // b128 index
            const uint64_t gaddr = (uint64_t)(uintptr_t)(x1 + i * 4);
            const uint32_t laddr = (uint32_t)(uintptr_t)(&sxyz[i * 4]);
            asm volatile("global_load_async_to_lds_b128 %0, %1, off"
                         :: "v"(laddr), "v"(gaddr) : "memory");
        }
        asm volatile("s_wait_asynccnt 0x0" ::: "memory");
    }
    __syncthreads();

    const int n = blockIdx.x * 256 + tid;
    const float* q = xyz2 + ((size_t)b * NN + n) * 3;
    const float qx = q[0], qy = q[1], qz = q[2];

    float d0 = 1e30f, d1 = 1e30f, d2 = 1e30f;
    int   i0 = 0,     i1 = 0,     i2 = 0;
    for (int s = 0; s < SS; ++s) {
        const float dx = qx - sxyz[s * 3 + 0];
        const float dy = qy - sxyz[s * 3 + 1];
        const float dz = qz - sxyz[s * 3 + 2];
        const float d  = dx * dx + dy * dy + dz * dz;
        if (d < d0)      { d2 = d1; i2 = i1; d1 = d0; i1 = i0; d0 = d; i0 = s; }
        else if (d < d1) { d2 = d1; i2 = i1; d1 = d;  i1 = s; }
        else if (d < d2) { d2 = d;  i2 = s; }
    }

    float w0 = 1.0f / fmaxf(d0, 1e-16f);
    float w1 = 1.0f / fmaxf(d1, 1e-16f);
    float w2 = 1.0f / fmaxf(d2, 1e-16f);
    const float inv = 1.0f / (w0 + w1 + w2);
    w0 *= inv; w1 *= inv; w2 *= inv;

    const float* f0 = points1 + ((size_t)b * SS + i0) * C1;
    const float* f1 = points1 + ((size_t)b * SS + i1) * C1;
    const float* f2 = points1 + ((size_t)b * SS + i2) * C1;

    const size_t row = (size_t)b * NN + n;
    __bf16* xr = X + row * CIN;

    #pragma unroll 4
    for (int c = 0; c < C1; c += 4) {
        const float4 a = *(const float4*)(f0 + c);
        const float4 g = *(const float4*)(f1 + c);
        const float4 h = *(const float4*)(f2 + c);
        xr[c + 0] = (__bf16)(w0 * a.x + w1 * g.x + w2 * h.x);
        xr[c + 1] = (__bf16)(w0 * a.y + w1 * g.y + w2 * h.y);
        xr[c + 2] = (__bf16)(w0 * a.z + w1 * g.z + w2 * h.z);
        xr[c + 3] = (__bf16)(w0 * a.w + w1 * g.w + w2 * h.w);
    }

    const float* p2 = points2 + row * C2;
    #pragma unroll 4
    for (int c = 0; c < C2; c += 4) {
        const float4 a = *(const float4*)(p2 + c);
        xr[C1 + c + 0] = (__bf16)a.x;
        xr[C1 + c + 1] = (__bf16)a.y;
        xr[C1 + c + 2] = (__bf16)a.z;
        xr[C1 + c + 3] = (__bf16)a.w;
    }
}

// ---------------------------------------------------------------------------
// 4) WMMA GEMM:  Y[M, HH] = Xb[M, K] * Wb[HH, K]^T + bias
//    Each wave computes a 16x64 strip: one A fragment feeds 4 WMMAs per
//    k-step (4x A reuse, ~12.8 flop/B from L2).  Block = 8 waves walking 8
//    consecutive row tiles, all sharing the same 64-column weight strip.
//    Fragment packing per CDNA5 16-bit A/B VGPR layout:
//      lane = hi*16 + l16 ; element j<8  -> k = kt + hi*8 + j
//                           element j>=8 -> k = kt + 16 + hi*8 + (j-8)
// ---------------------------------------------------------------------------
template <int KDIM>
__global__ __launch_bounds__(256) void gemm_bias_kernel(
    const __bf16* __restrict__ Xb,    // [M, KDIM]
    const __bf16* __restrict__ Wb,    // [HH, KDIM]  (row-major -> B^T rows)
    const float*  __restrict__ bias,  // [HH]
    float* __restrict__ Y)            // [M, HH]
{
    const int wave = threadIdx.x >> 5;
    const int lane = threadIdx.x & 31;
    const int half = lane >> 4;     // 0 or 1
    const int l16  = lane & 15;

    const int mtiles = MM >> 4;                 // 4096 row tiles
    const int tile   = blockIdx.x * 8 + wave;   // wave-tile id
    const int ng     = tile / mtiles;           // 64-col group: 0..3
    const int mt     = tile - ng * mtiles;      // row tile

    const __bf16* arow  = Xb + (size_t)(mt * 16 + l16) * KDIM;
    const __bf16* brow0 = Wb + (size_t)(ng * 64 + l16) * KDIM;

    v8f acc[4] = {};
    #pragma unroll
    for (int kt = 0; kt < KDIM; kt += 32) {
        const int k0 = kt + half * 8;
        const v8bf alo = *(const v8bf*)(arow + k0);
        const v8bf ahi = *(const v8bf*)(arow + k0 + 16);
        if (kt + 32 < KDIM) __builtin_prefetch(arow + k0 + 32, 0, 1);
        v16bf a;
        #pragma unroll
        for (int j = 0; j < 8; ++j) { a[j] = alo[j]; a[j + 8] = ahi[j]; }

        #pragma unroll
        for (int t = 0; t < 4; ++t) {
            const __bf16* br = brow0 + (size_t)(t * 16) * KDIM;
            const v8bf blo = *(const v8bf*)(br + k0);
            const v8bf bhi = *(const v8bf*)(br + k0 + 16);
            v16bf bfr;
            #pragma unroll
            for (int j = 0; j < 8; ++j) { bfr[j] = blo[j]; bfr[j + 8] = bhi[j]; }
            acc[t] = __builtin_amdgcn_wmma_f32_16x16x32_bf16(
                /*neg_a=*/false, a, /*neg_b=*/false, bfr,
                /*c_mod=*/(short)0, acc[t], /*reuse_a=*/false, /*reuse_b=*/false);
        }
    }

    // D layout: VGPR r -> row (mt*16 + half*8 + r), col = ng*64 + t*16 + l16
    #pragma unroll
    for (int t = 0; t < 4; ++t) {
        const int   col  = ng * 64 + t * 16 + l16;
        const float bval = bias[col];
        #pragma unroll
        for (int r = 0; r < 8; ++r) {
            Y[(size_t)(mt * 16 + half * 8 + r) * HH + col] = acc[t][r] + bval;
        }
    }
}

// ---------------------------------------------------------------------------
// 5) BN batch stats: per-channel sum / sumsq over M rows (coalesced columns,
//    block-local accumulation, two f32 atomics per thread per block).
//    grid = M/256, block = 256 (thread == channel).
// ---------------------------------------------------------------------------
__global__ __launch_bounds__(256) void bn_stats_kernel(
    const float* __restrict__ Y,      // [M, 256]
    float* __restrict__ sum,          // [256]
    float* __restrict__ sumsq)        // [256]
{
    const int ch   = threadIdx.x;
    const int row0 = blockIdx.x * 256;
    float s = 0.0f, ss = 0.0f;
    #pragma unroll 4
    for (int i = 0; i < 256; ++i) {
        const float v = Y[(size_t)(row0 + i) * 256 + ch];
        s += v;
        ss += v * v;
    }
    atomicAdd(&sum[ch], s);
    atomicAdd(&sumsq[ch], ss);
}

// ---------------------------------------------------------------------------
// 6) BN normalize + affine + ReLU, emit bf16 (feeds next GEMM)
// ---------------------------------------------------------------------------
__global__ __launch_bounds__(256) void bn_relu_bf16_kernel(
    const float* __restrict__ Y, const float* __restrict__ sum,
    const float* __restrict__ sumsq, const float* __restrict__ g,
    const float* __restrict__ be, __bf16* __restrict__ out)
{
    const size_t idx = (size_t)blockIdx.x * 256 + threadIdx.x;
    const int ch = (int)(idx & 255);
    const float mean = sum[ch] * (1.0f / MM);
    const float var  = sumsq[ch] * (1.0f / MM) - mean * mean;
    float v = (Y[idx] - mean) * rsqrtf(var + BN_EPS) * g[ch] + be[ch];
    v = fmaxf(v, 0.0f);
    out[idx] = (__bf16)v;
}

// ---------------------------------------------------------------------------
// 7) BN normalize + affine + ReLU, emit f32 (final output [B,N,H])
// ---------------------------------------------------------------------------
__global__ __launch_bounds__(256) void bn_relu_f32_kernel(
    const float* __restrict__ Y, const float* __restrict__ sum,
    const float* __restrict__ sumsq, const float* __restrict__ g,
    const float* __restrict__ be, float* __restrict__ out)
{
    const size_t idx = (size_t)blockIdx.x * 256 + threadIdx.x;
    const int ch = (int)(idx & 255);
    const float mean = sum[ch] * (1.0f / MM);
    const float var  = sumsq[ch] * (1.0f / MM) - mean * mean;
    float v = (Y[idx] - mean) * rsqrtf(var + BN_EPS) * g[ch] + be[ch];
    out[idx] = fmaxf(v, 0.0f);
}

// ---------------------------------------------------------------------------
// Launcher
// ---------------------------------------------------------------------------
extern "C" void kernel_launch(void* const* d_in, const int* in_sizes, int n_in,
                              void* d_out, int out_size, void* d_ws, size_t ws_size,
                              hipStream_t stream) {
    (void)in_sizes; (void)n_in; (void)out_size; (void)ws_size;

    const float* points1 = (const float*)d_in[0];   // [B,S,C1]
    const float* points2 = (const float*)d_in[1];   // [B,N,C2]
    const float* xyz1    = (const float*)d_in[2];   // [B,S,3]
    const float* xyz2    = (const float*)d_in[3];   // [B,N,3]
    const float* w1      = (const float*)d_in[4];   // [H, CIN]
    const float* b1      = (const float*)d_in[5];
    const float* g1      = (const float*)d_in[6];
    const float* be1     = (const float*)d_in[7];
    const float* w2      = (const float*)d_in[8];   // [H, H]
    const float* b2      = (const float*)d_in[9];
    const float* g2      = (const float*)d_in[10];
    const float* be2     = (const float*)d_in[11];
    float* out = (float*)d_out;

    // Workspace carve-up (256B aligned regions)
    char* ws = (char*)d_ws;
    size_t off = 0;
    auto take = [&](size_t bytes) {
        void* p = ws + off;
        off += (bytes + 255) & ~(size_t)255;
        return p;
    };
    __bf16* W1bf = (__bf16*)take((size_t)HH * CIN * sizeof(__bf16));  // 192 KB
    __bf16* W2bf = (__bf16*)take((size_t)HH * HH  * sizeof(__bf16));  // 128 KB
    __bf16* X    = (__bf16*)take((size_t)MM * CIN * sizeof(__bf16));  // 48 MB
    float*  Y    = (float*) take((size_t)MM * HH  * sizeof(float));   // 64 MB
    __bf16* X2   = (__bf16*)take((size_t)MM * HH  * sizeof(__bf16));  // 32 MB
    float*  st   = (float*) take(512 * sizeof(float));                // sum | sumsq
    float*  s_sum = st;
    float*  s_sq  = st + 256;

    // 1) weights -> bf16
    f32_to_bf16_kernel<<<(HH * CIN + 255) / 256, 256, 0, stream>>>(w1, W1bf, HH * CIN);
    f32_to_bf16_kernel<<<(HH * HH + 255) / 256, 256, 0, stream>>>(w2, W2bf, HH * HH);

    // 2) KNN interpolate + concat -> X bf16
    knn_build_x_kernel<<<dim3(NN / 256, BB), 256, 0, stream>>>(points1, points2, xyz1, xyz2, X);

    // 3) GEMM1 (K=384): Y = X * W1^T + b1
    {
        const int wave_tiles = (MM / 16) * (HH / 64);   // 16384
        gemm_bias_kernel<CIN><<<wave_tiles / 8, 256, 0, stream>>>(X, W1bf, b1, Y);
    }

    // 4) BN1 stats
    zero_f32_kernel<<<2, 256, 0, stream>>>(st, 512);
    bn_stats_kernel<<<MM / 256, 256, 0, stream>>>(Y, s_sum, s_sq);

    // 5) BN1 + ReLU -> X2 bf16
    bn_relu_bf16_kernel<<<(MM * HH) / 256, 256, 0, stream>>>(Y, s_sum, s_sq, g1, be1, X2);

    // 6) GEMM2 (K=256): Y = X2 * W2^T + b2   (reuse Y)
    {
        const int wave_tiles = (MM / 16) * (HH / 64);
        gemm_bias_kernel<HH><<<wave_tiles / 8, 256, 0, stream>>>(X2, W2bf, b2, Y);
    }

    // 7) BN2 stats
    zero_f32_kernel<<<2, 256, 0, stream>>>(st, 512);
    bn_stats_kernel<<<MM / 256, 256, 0, stream>>>(Y, s_sum, s_sq);

    // 8) BN2 + ReLU -> d_out f32 [B,N,H]
    bn_relu_f32_kernel<<<(MM * HH) / 256, 256, 0, stream>>>(Y, s_sum, s_sq, g2, be2, out);
}
